// TemplatePairBlockWrapper_81604378624203
// MI455X (gfx1250) — compile-verified, matrix operand order checked by reading, not run on hardware
//
#include <hip/hip_runtime.h>
#include <hip/hip_bf16.h>
#include <cstdint>
#include <cstddef>

// ---------------------------------------------------------------------------
// Dims (from reference): B=1, T=2, N=192, C=64, CH_MUL=64, H=4, CH_ATT=16
// ---------------------------------------------------------------------------
static constexpr int Nn  = 192;
static constexpr int NNp = Nn * Nn;           // 36864 positions per T
static constexpr int TNN = 2 * NNp;           // 73728 rows total
static constexpr int TILES16 = TNN / 16;      // 4608 row tiles
static constexpr int ROWBLKS = TILES16 / 4;   // 1152 blocks of 4 waves

typedef __attribute__((ext_vector_type(16))) __bf16 v16bf;
typedef __attribute__((ext_vector_type(8)))  __bf16 v8bf;
typedef __attribute__((ext_vector_type(8)))  float  v8f;

__device__ __forceinline__ v8f wmma_bf16(v16bf a, v16bf b, v8f c) {
  return __builtin_amdgcn_wmma_f32_16x16x32_bf16(false, a, false, b, (short)0, c, false, false);
}
__device__ __forceinline__ v16bf cat8(v8bf lo, v8bf hi) {
  return __builtin_shufflevector(lo, hi, 0, 1, 2, 3, 4, 5, 6, 7,
                                 8, 9, 10, 11, 12, 13, 14, 15);
}

// Fragment-major swizzle: element (mn, kk) of a 16x32 (or 32x16) tile goes to
// bf16 offset sw_idx within a v16bf[32] chunk so that lane L reads its whole
// fragment as one contiguous 32B vector (2 x ds_load_b128 / global_load_b128).
// Derived from ISA 7.12.2 16-bit A/B layout: k = (e>=8?16:0) + half*8 + (e&7).
__device__ __forceinline__ int sw_idx(int mn, int kk) {
  int half = (kk >> 3) & 1;
  int e = (kk & 7) | ((kk >> 4) << 3);
  return ((mn + (half << 4)) << 4) + e;
}

__device__ __forceinline__ float sigmoidf_(float x) { return 1.f / (1.f + __expf(-x)); }

// Wave-cooperative LayerNorm over 64 channels: lane supplies ch lane and lane+32.
__device__ __forceinline__ void ln64(const float* __restrict__ row,
                                     const float* __restrict__ g,
                                     const float* __restrict__ b,
                                     int lane, float& y0, float& y1) {
  float x0 = row[lane], x1 = row[lane + 32];
  float s = x0 + x1;
#pragma unroll
  for (int m = 16; m >= 1; m >>= 1) s += __shfl_xor(s, m, 32);
  float mu = s * (1.f / 64.f);
  float d0 = x0 - mu, d1 = x1 - mu;
  float v = d0 * d0 + d1 * d1;
#pragma unroll
  for (int m = 16; m >= 1; m >>= 1) v += __shfl_xor(v, m, 32);
  float rs = rsqrtf(v * (1.f / 64.f) + 1e-5f);
  y0 = d0 * rs * g[lane] + b[lane];
  y1 = d1 * rs * g[lane + 32] + b[lane + 32];
}

struct MulP {
  const float *ln_g, *ln_b;
  const float *ag_w, *ag_b, *ap_w, *ap_b, *bg_w, *bg_b, *bp_w, *bp_b;
  const float *g_w, *g_b, *o_w, *o_b, *ln_out_g, *ln_out_b;
};
struct AttP {
  const float *ln_g, *ln_b, *q_w, *k_w, *v_w, *tb_w, *g_w, *g_b, *o_w, *o_b;
};
struct TransP { const float *ln_g, *ln_b, *w1, *b1, *w2, *b2; };

// ---------------------------------------------------------------------------
// Triangle-mult projections. Stores a/b channel-major (optionally transposed
// so the einsum always uses the contiguous-k "outgoing" access pattern).
// ---------------------------------------------------------------------------
__global__ __launch_bounds__(128) void k_mul_proj(const float* __restrict__ tsrc,
                                                  const float* __restrict__ mask,
                                                  MulP p, __bf16* a_cm, __bf16* b_cm,
                                                  float* g_out, int store_trans) {
  __shared__ v16bf Wl[5][4][2][32];   // [mat][ct][kchunk][lane]  40KB
  __shared__ v16bf Zl[4][2][32];      // [wave][kchunk][lane]      8KB
  int tid = threadIdx.x, lane = tid & 31, wv = tid >> 5;
  const float* Ws[5] = { p.ag_w, p.ap_w, p.bg_w, p.bp_w, p.g_w };
  for (int w = 0; w < 5; ++w)
    for (int idx = tid; idx < 64 * 64; idx += blockDim.x) {
      int k = idx >> 6, n = idx & 63;
      ((__bf16*)Wl[w][n >> 4][k >> 5])[sw_idx(n & 15, k & 31)] = (__bf16)Ws[w][idx];
    }
  __syncthreads();

  int tile = blockIdx.x * 4 + wv, row0 = tile * 16;
  __bf16* Z0 = (__bf16*)Zl[wv][0];
  __bf16* Z1 = (__bf16*)Zl[wv][1];
  for (int rr = 0; rr < 16; ++rr) {
    float y0, y1;
    ln64(tsrc + (size_t)(row0 + rr) * 64, p.ln_g, p.ln_b, lane, y0, y1);
    Z0[sw_idx(rr, lane)] = (__bf16)y0;
    Z1[sw_idx(rr, lane)] = (__bf16)y1;
  }
  v16bf a0 = Zl[wv][0][lane], a1 = Zl[wv][1][lane];
  int half = lane >> 4, nn = lane & 15;
  for (int ct = 0; ct < 4; ++ct) {
    v8f acc[5];
#pragma unroll
    for (int w = 0; w < 5; ++w) {
      v8f c = {};
      c = wmma_bf16(a0, Wl[w][ct][0][lane], c);
      c = wmma_bf16(a1, Wl[w][ct][1][lane], c);
      acc[w] = c;
    }
    int n = ct * 16 + nn;
#pragma unroll
    for (int r = 0; r < 8; ++r) {
      int m = r + half * 8, grow = row0 + m;
      int tix = grow / NNp, rem = grow % NNp, i = rem / Nn, j = rem % Nn;
      float mv = mask[grow];
      float av = mv * sigmoidf_(acc[0][r] + p.ag_b[n]) * (acc[1][r] + p.ap_b[n]);
      float bv = mv * sigmoidf_(acc[2][r] + p.bg_b[n]) * (acc[3][r] + p.bp_b[n]);
      size_t pos = store_trans ? ((size_t)j * Nn + i) : ((size_t)i * Nn + j);
      size_t cm = ((size_t)tix * 64 + n) * NNp + pos;
      a_cm[cm] = (__bf16)av;
      b_cm[cm] = (__bf16)bv;
      g_out[(size_t)grow * 64 + n] = sigmoidf_(acc[4][r] + p.g_b[n]);
    }
  }
}

// ---------------------------------------------------------------------------
// Triangle einsum: 64 independent 192x192x192 GEMMs per T slice.
// x[i,j,c] = sum_k A'[i,k] B'[j,k] on channel-major (pre-transposed) operands.
// ---------------------------------------------------------------------------
__global__ __launch_bounds__(256) void k_einsum(const __bf16* __restrict__ a_cm,
                                                const __bf16* __restrict__ b_cm,
                                                float* __restrict__ x_f) {
  int lane = threadIdx.x & 31, wv = threadIdx.x >> 5;
  int blk = blockIdx.x;
  int jt = blk % 12, it = (blk / 12) % 12, tix = blk / 144;
  int i0 = it * 16, j0 = jt * 16;
  size_t base = (size_t)tix * NNp;
  int half = lane >> 4, nn = lane & 15;
  for (int cc = 0; cc < 8; ++cc) {
    int c = wv * 8 + cc;
    const __bf16* A = a_cm + ((size_t)tix * 64 + c) * NNp + (size_t)(i0 + nn) * Nn + half * 8;
    const __bf16* B = b_cm + ((size_t)tix * 64 + c) * NNp + (size_t)(j0 + nn) * Nn + half * 8;
    v8f acc = {};
    for (int kc = 0; kc < 6; ++kc) {
      int k0 = kc * 32;
      v16bf fa = cat8(*(const v8bf*)(A + k0), *(const v8bf*)(A + k0 + 16));
      v16bf fb = cat8(*(const v8bf*)(B + k0), *(const v8bf*)(B + k0 + 16));
      acc = wmma_bf16(fa, fb, acc);
    }
#pragma unroll
    for (int r = 0; r < 8; ++r) {
      int m = r + half * 8;
      x_f[(base + (size_t)(i0 + m) * Nn + (j0 + nn)) * 64 + c] = acc[r];
    }
  }
}

// ---------------------------------------------------------------------------
// Triangle-mult epilogue: t += g * (LN(x) @ o_w + o_b)
// ---------------------------------------------------------------------------
__global__ __launch_bounds__(128) void k_mul_out(float* __restrict__ t_io,
                                                 const float* __restrict__ x_f,
                                                 const float* __restrict__ g_f, MulP p) {
  __shared__ v16bf Wo[4][2][32];
  __shared__ v16bf Zl[4][2][32];
  int tid = threadIdx.x, lane = tid & 31, wv = tid >> 5;
  for (int idx = tid; idx < 64 * 64; idx += blockDim.x) {
    int k = idx >> 6, n = idx & 63;
    ((__bf16*)Wo[n >> 4][k >> 5])[sw_idx(n & 15, k & 31)] = (__bf16)p.o_w[idx];
  }
  __syncthreads();
  int tile = blockIdx.x * 4 + wv, row0 = tile * 16;
  __bf16* Z0 = (__bf16*)Zl[wv][0];
  __bf16* Z1 = (__bf16*)Zl[wv][1];
  for (int rr = 0; rr < 16; ++rr) {
    float y0, y1;
    ln64(x_f + (size_t)(row0 + rr) * 64, p.ln_out_g, p.ln_out_b, lane, y0, y1);
    Z0[sw_idx(rr, lane)] = (__bf16)y0;
    Z1[sw_idx(rr, lane)] = (__bf16)y1;
  }
  v16bf a0 = Zl[wv][0][lane], a1 = Zl[wv][1][lane];
  int half = lane >> 4, nn = lane & 15;
  for (int ct = 0; ct < 4; ++ct) {
    v8f c = {};
    c = wmma_bf16(a0, Wo[ct][0][lane], c);
    c = wmma_bf16(a1, Wo[ct][1][lane], c);
    int n = ct * 16 + nn;
#pragma unroll
    for (int r = 0; r < 8; ++r) {
      int m = r + half * 8;
      size_t idx = (size_t)(row0 + m) * 64 + n;
      t_io[idx] += g_f[idx] * (c[r] + p.o_b[n]);
    }
  }
}

// ---------------------------------------------------------------------------
// Attention projections: q*scale, k (row-major bf16), v (channel-major bf16),
// g (sigmoid f32), tb (f32).  transposed!=0 reads t at (j,i).
// ---------------------------------------------------------------------------
__global__ __launch_bounds__(128) void k_att_proj(const float* __restrict__ tsrc, AttP p,
                                                  int transposed, __bf16* q_o, __bf16* k_o,
                                                  __bf16* v_cm, float* g_o, float* tb_o) {
  __shared__ v16bf Wl[4][4][2][32];   // 32KB
  __shared__ v16bf Zl[4][2][32];
  int tid = threadIdx.x, lane = tid & 31, wv = tid >> 5;
  const float* Ws[4] = { p.q_w, p.k_w, p.v_w, p.g_w };
  for (int w = 0; w < 4; ++w)
    for (int idx = tid; idx < 64 * 64; idx += blockDim.x) {
      int k = idx >> 6, n = idx & 63;
      ((__bf16*)Wl[w][n >> 4][k >> 5])[sw_idx(n & 15, k & 31)] = (__bf16)Ws[w][idx];
    }
  __syncthreads();

  int tile = blockIdx.x * 4 + wv, row0 = tile * 16;
  __bf16* Z0 = (__bf16*)Zl[wv][0];
  __bf16* Z1 = (__bf16*)Zl[wv][1];
  for (int rr = 0; rr < 16; ++rr) {
    int grow = row0 + rr;
    int tix = grow / NNp, rem = grow % NNp, i = rem / Nn, j = rem % Nn;
    size_t src = transposed ? ((size_t)tix * NNp + (size_t)j * Nn + i) : (size_t)grow;
    float y0, y1;
    ln64(tsrc + src * 64, p.ln_g, p.ln_b, lane, y0, y1);
    Z0[sw_idx(rr, lane)] = (__bf16)y0;
    Z1[sw_idx(rr, lane)] = (__bf16)y1;
#pragma unroll
    for (int h = 0; h < 4; ++h) {   // tb[pos,h] = sum_c zl*tb_w[c,h]
      float tv = y0 * p.tb_w[lane * 4 + h] + y1 * p.tb_w[(lane + 32) * 4 + h];
#pragma unroll
      for (int mm = 16; mm >= 1; mm >>= 1) tv += __shfl_xor(tv, mm, 32);
      if (lane == 0) tb_o[(size_t)grow * 4 + h] = tv;
    }
  }
  v16bf a0 = Zl[wv][0][lane], a1 = Zl[wv][1][lane];
  int half = lane >> 4, nn = lane & 15;
  for (int ct = 0; ct < 4; ++ct) {
    v8f acc[4];
#pragma unroll
    for (int w = 0; w < 4; ++w) {
      v8f c = {};
      c = wmma_bf16(a0, Wl[w][ct][0][lane], c);
      c = wmma_bf16(a1, Wl[w][ct][1][lane], c);
      acc[w] = c;
    }
    int n = ct * 16 + nn;
#pragma unroll
    for (int r = 0; r < 8; ++r) {
      int m = r + half * 8, grow = row0 + m;
      size_t idx = (size_t)grow * 64 + n;
      q_o[idx] = (__bf16)(acc[0][r] * 0.25f);     // CH_ATT^-0.5
      k_o[idx] = (__bf16)acc[1][r];
      v_cm[(size_t)n * TNN + grow] = (__bf16)acc[2][r];
      g_o[idx] = sigmoidf_(acc[3][r] + p.g_b[n]);
    }
  }
}

// ---------------------------------------------------------------------------
// Attention core: scores = Q K^T + tb + maskbias, softmax over kk, O = P V.
// ---------------------------------------------------------------------------
__global__ __launch_bounds__(128) void k_att_core(const __bf16* __restrict__ q_b,
                                                  const __bf16* __restrict__ k_b,
                                                  const __bf16* __restrict__ v_cm,
                                                  const float* __restrict__ g_f,
                                                  const float* __restrict__ tb_f,
                                                  const float* __restrict__ mask,
                                                  int transposed, float* __restrict__ o_f) {
  __shared__ v16bf Pl[4][6][32];      // [wave][kchunk][lane]  24KB
  int lane = threadIdx.x & 31, wv = threadIdx.x >> 5;
  int task = blockIdx.x * 4 + wv;
  int qt = task % 12;
  int i  = (task / 12) % Nn;
  int h  = (task / (12 * Nn)) % 4;
  int tix = task / (12 * Nn * 4);
  size_t rowbase = (size_t)tix * NNp + (size_t)i * Nn;
  int half = lane >> 4, nn = lane & 15;
  int q0 = qt * 16;
  v8bf z8 = {};

  // Q fragment: K=16 real (always k<16 in low run), high run zero-padded
  v16bf fq = cat8(*(const v8bf*)(q_b + (rowbase + q0 + nn) * 64 + h * 16 + half * 8), z8);

  float s[12][8];
  for (int kt = 0; kt < 12; ++kt) {
    v16bf fk = cat8(*(const v8bf*)(k_b + (rowbase + kt * 16 + nn) * 64 + h * 16 + half * 8), z8);
    v8f acc = {};
    acc = wmma_bf16(fq, fk, acc);
    int kk = kt * 16 + nn;
    float mv = transposed ? mask[(size_t)tix * NNp + (size_t)kk * Nn + i]
                          : mask[(size_t)tix * NNp + (size_t)i * Nn + kk];
    float mb = (mv - 1.f) * 1e9f;
#pragma unroll
    for (int r = 0; r < 8; ++r) {
      int m = r + half * 8;
      s[kt][r] = acc[r] + mb
               + tb_f[((size_t)tix * NNp + (size_t)(q0 + m) * Nn + kk) * 4 + h];
    }
  }
#pragma unroll
  for (int r = 0; r < 8; ++r) {
    float mx = s[0][r];
#pragma unroll
    for (int kt = 1; kt < 12; ++kt) mx = fmaxf(mx, s[kt][r]);
#pragma unroll
    for (int mmk = 8; mmk >= 1; mmk >>= 1) mx = fmaxf(mx, __shfl_xor(mx, mmk, 32));
    float sum = 0.f;
#pragma unroll
    for (int kt = 0; kt < 12; ++kt) { s[kt][r] = __expf(s[kt][r] - mx); sum += s[kt][r]; }
#pragma unroll
    for (int mmk = 8; mmk >= 1; mmk >>= 1) sum += __shfl_xor(sum, mmk, 32);
    float inv = 1.f / sum;
#pragma unroll
    for (int kt = 0; kt < 12; ++kt) s[kt][r] *= inv;
  }
  // stash P swizzled so each A-fragment is one contiguous 32B vector
#pragma unroll
  for (int kt = 0; kt < 12; ++kt)
#pragma unroll
    for (int r = 0; r < 8; ++r) {
      int m = r + half * 8, kk = kt * 16 + nn;
      ((__bf16*)Pl[wv][kk >> 5])[sw_idx(m, kk & 31)] = (__bf16)s[kt][r];
    }
  // O = P @ V  (K = 192 in 6 chunks), V channel-major -> contiguous k runs
  const __bf16* vb = v_cm + (size_t)(h * 16 + nn) * TNN + rowbase + half * 8;
  v8f acc = {};
  for (int kc = 0; kc < 6; ++kc) {
    v16bf fp = Pl[wv][kc][lane];
    v16bf fv = cat8(*(const v8bf*)(vb + kc * 32), *(const v8bf*)(vb + kc * 32 + 16));
    acc = wmma_bf16(fp, fv, acc);
  }
#pragma unroll
  for (int r = 0; r < 8; ++r) {
    int m = r + half * 8;
    size_t idx = (rowbase + q0 + m) * 64 + h * 16 + nn;
    o_f[idx] = g_f[idx] * acc[r];
  }
}

// ---------------------------------------------------------------------------
// Attention epilogue: t[dest] += (o @ o_w + o_b), dest swapped if transposed.
// ---------------------------------------------------------------------------
__global__ __launch_bounds__(128) void k_att_out(float* __restrict__ t_io,
                                                 const float* __restrict__ o_f,
                                                 AttP p, int transposed) {
  __shared__ v16bf Wo[4][2][32];
  __shared__ v16bf Zl[4][2][32];
  int tid = threadIdx.x, lane = tid & 31, wv = tid >> 5;
  for (int idx = tid; idx < 64 * 64; idx += blockDim.x) {
    int k = idx >> 6, n = idx & 63;
    ((__bf16*)Wo[n >> 4][k >> 5])[sw_idx(n & 15, k & 31)] = (__bf16)p.o_w[idx];
  }
  __syncthreads();
  int tile = blockIdx.x * 4 + wv, row0 = tile * 16;
  __bf16* Z0 = (__bf16*)Zl[wv][0];
  __bf16* Z1 = (__bf16*)Zl[wv][1];
  for (int rr = 0; rr < 16; ++rr) {
    const float* src = o_f + (size_t)(row0 + rr) * 64;
    Z0[sw_idx(rr, lane)] = (__bf16)src[lane];
    Z1[sw_idx(rr, lane)] = (__bf16)src[lane + 32];
  }
  v16bf a0 = Zl[wv][0][lane], a1 = Zl[wv][1][lane];
  int half = lane >> 4, nn = lane & 15;
  for (int ct = 0; ct < 4; ++ct) {
    v8f c = {};
    c = wmma_bf16(a0, Wo[ct][0][lane], c);
    c = wmma_bf16(a1, Wo[ct][1][lane], c);
    int n = ct * 16 + nn;
#pragma unroll
    for (int r = 0; r < 8; ++r) {
      int m = r + half * 8, grow = row0 + m;
      int tix = grow / NNp, rem = grow % NNp, ii = rem / Nn, jj = rem % Nn;
      size_t dest = transposed ? ((size_t)tix * NNp + (size_t)jj * Nn + ii) : (size_t)grow;
      t_io[dest * 64 + n] += c[r] + p.o_b[n];
    }
  }
}

// ---------------------------------------------------------------------------
// Transition: t += mask * (relu(LN(t)@w1+b1) @ w2 + b2)   (64 -> 128 -> 64)
// ---------------------------------------------------------------------------
__global__ __launch_bounds__(128) void k_trans(float* __restrict__ t_io,
                                               const float* __restrict__ mask, TransP p) {
  __shared__ v16bf W1[8][2][32];      // 16KB  (64 -> 128)
  __shared__ v16bf W2[4][4][32];      // 16KB  (128 -> 64)
  __shared__ v16bf Zl[4][2][32];      //  8KB
  __shared__ v16bf Hl[4][4][32];      // 16KB  hidden, swizzled for 2nd GEMM
  int tid = threadIdx.x, lane = tid & 31, wv = tid >> 5;
  for (int idx = tid; idx < 64 * 128; idx += blockDim.x) {
    int k = idx >> 7, n = idx & 127;
    ((__bf16*)W1[n >> 4][k >> 5])[sw_idx(n & 15, k & 31)] = (__bf16)p.w1[idx];
  }
  for (int idx = tid; idx < 128 * 64; idx += blockDim.x) {
    int k = idx >> 6, n = idx & 63;
    ((__bf16*)W2[n >> 4][k >> 5])[sw_idx(n & 15, k & 31)] = (__bf16)p.w2[idx];
  }
  __syncthreads();
  int tile = blockIdx.x * 4 + wv, row0 = tile * 16;
  __bf16* Z0 = (__bf16*)Zl[wv][0];
  __bf16* Z1 = (__bf16*)Zl[wv][1];
  for (int rr = 0; rr < 16; ++rr) {
    float y0, y1;
    ln64(t_io + (size_t)(row0 + rr) * 64, p.ln_g, p.ln_b, lane, y0, y1);
    Z0[sw_idx(rr, lane)] = (__bf16)y0;
    Z1[sw_idx(rr, lane)] = (__bf16)y1;
  }
  v16bf a0 = Zl[wv][0][lane], a1 = Zl[wv][1][lane];
  int half = lane >> 4, nn = lane & 15;
  for (int ct = 0; ct < 8; ++ct) {        // h = relu(zl@w1 + b1)
    v8f c = {};
    c = wmma_bf16(a0, W1[ct][0][lane], c);
    c = wmma_bf16(a1, W1[ct][1][lane], c);
    int n = ct * 16 + nn;
#pragma unroll
    for (int r = 0; r < 8; ++r) {
      int m = r + half * 8;
      ((__bf16*)Hl[wv][n >> 5])[sw_idx(m, n & 31)] = (__bf16)fmaxf(c[r] + p.b1[n], 0.f);
    }
  }
  for (int ct = 0; ct < 4; ++ct) {        // out = h@w2 + b2, K=128
    v8f c = {};
#pragma unroll
    for (int kc = 0; kc < 4; ++kc)
      c = wmma_bf16(Hl[wv][kc][lane], W2[ct][kc][lane], c);
    int n = ct * 16 + nn;
#pragma unroll
    for (int r = 0; r < 8; ++r) {
      int m = r + half * 8, grow = row0 + m;
      t_io[(size_t)grow * 64 + n] += (c[r] + p.b2[n]) * mask[grow];
    }
  }
}

// ---------------------------------------------------------------------------
// Host-side unpack + launch.  Layout assumption: d_in[0]=t, d_in[1]=mask, then
// param leaves pytree-flattened (alphabetical at each dict level):
//   att_end(10), att_start(10), mul_in(16), mul_out(16), trans(6)
// ---------------------------------------------------------------------------
static void fill_att(AttP& a, void* const* d, int b) {
  a.g_b = (const float*)d[b + 0]; a.g_w = (const float*)d[b + 1];
  a.k_w = (const float*)d[b + 2]; a.ln_b = (const float*)d[b + 3];
  a.ln_g = (const float*)d[b + 4]; a.o_b = (const float*)d[b + 5];
  a.o_w = (const float*)d[b + 6]; a.q_w = (const float*)d[b + 7];
  a.tb_w = (const float*)d[b + 8]; a.v_w = (const float*)d[b + 9];
}
static void fill_mul(MulP& m, void* const* d, int b) {
  m.ag_b = (const float*)d[b + 0];  m.ag_w = (const float*)d[b + 1];
  m.ap_b = (const float*)d[b + 2];  m.ap_w = (const float*)d[b + 3];
  m.bg_b = (const float*)d[b + 4];  m.bg_w = (const float*)d[b + 5];
  m.bp_b = (const float*)d[b + 6];  m.bp_w = (const float*)d[b + 7];
  m.g_b  = (const float*)d[b + 8];  m.g_w  = (const float*)d[b + 9];
  m.ln_b = (const float*)d[b + 10]; m.ln_g = (const float*)d[b + 11];
  m.ln_out_b = (const float*)d[b + 12]; m.ln_out_g = (const float*)d[b + 13];
  m.o_b  = (const float*)d[b + 14]; m.o_w  = (const float*)d[b + 15];
}
static void fill_trans(TransP& t, void* const* d, int b) {
  t.b1 = (const float*)d[b + 0]; t.b2 = (const float*)d[b + 1];
  t.ln_b = (const float*)d[b + 2]; t.ln_g = (const float*)d[b + 3];
  t.w1 = (const float*)d[b + 4]; t.w2 = (const float*)d[b + 5];
}

extern "C" void kernel_launch(void* const* d_in, const int* in_sizes, int n_in,
                              void* d_out, int out_size, void* d_ws, size_t ws_size,
                              hipStream_t stream) {
  (void)in_sizes; (void)n_in; (void)out_size; (void)ws_size;
  const float* t_in = (const float*)d_in[0];
  const float* mask = (const float*)d_in[1];
  AttP att_end, att_start; MulP mul_in, mul_out; TransP trans;
  fill_att(att_end, d_in, 2);
  fill_att(att_start, d_in, 12);
  fill_mul(mul_in, d_in, 22);
  fill_mul(mul_out, d_in, 38);
  fill_trans(trans, d_in, 54);

  char* ws = (char*)d_ws;
  auto take = [&](size_t bytes) { char* p = ws; ws += (bytes + 255) & ~(size_t)255; return p; };
  const size_t ELEMS = (size_t)TNN * 64;
  __bf16* bf0 = (__bf16*)take(ELEMS * 2);        // mul a  / att q
  __bf16* bf1 = (__bf16*)take(ELEMS * 2);        // mul b  / att k
  __bf16* bf2 = (__bf16*)take(ELEMS * 2);        // att v (channel-major)
  float*  g_f = (float*)take(ELEMS * 4);         // gates
  float*  x_f = (float*)take(ELEMS * 4);         // einsum out / att o
  float*  tb_f = (float*)take((size_t)TNN * 4 * 4);

  float* t = (float*)d_out;                      // evolving pair activations
  hipMemcpyAsync(t, t_in, ELEMS * 4, hipMemcpyDeviceToDevice, stream);

  dim3 rb(128), rg(ROWBLKS);
  // ---- triangle mult, outgoing ----
  k_mul_proj<<<rg, rb, 0, stream>>>(t, mask, mul_out, bf0, bf1, g_f, 0);
  k_einsum<<<288, 256, 0, stream>>>(bf0, bf1, x_f);
  k_mul_out<<<rg, rb, 0, stream>>>(t, x_f, g_f, mul_out);
  // ---- triangle mult, incoming (producer stores transposed) ----
  k_mul_proj<<<rg, rb, 0, stream>>>(t, mask, mul_in, bf0, bf1, g_f, 1);
  k_einsum<<<288, 256, 0, stream>>>(bf0, bf1, x_f);
  k_mul_out<<<rg, rb, 0, stream>>>(t, x_f, g_f, mul_in);
  // ---- triangle attention, starting node ----
  k_att_proj<<<rg, rb, 0, stream>>>(t, att_start, 0, bf0, bf1, bf2, g_f, tb_f);
  k_att_core<<<4608, 128, 0, stream>>>(bf0, bf1, bf2, g_f, tb_f, mask, 0, x_f);
  k_att_out<<<rg, rb, 0, stream>>>(t, x_f, att_start, 0);
  // ---- triangle attention, ending node (index-transposed) ----
  k_att_proj<<<rg, rb, 0, stream>>>(t, att_end, 1, bf0, bf1, bf2, g_f, tb_f);
  k_att_core<<<4608, 128, 0, stream>>>(bf0, bf1, bf2, g_f, tb_f, mask, 1, x_f);
  k_att_out<<<rg, rb, 0, stream>>>(t, x_f, att_end, 1);
  // ---- pair transition ----
  k_trans<<<rg, rb, 0, stream>>>(t, mask, trans);
}
